// GAT_GCN_45981919871648
// MI455X (gfx1250) — compile-verified, matrix operand order checked by reading, not run on hardware
//
// GAT_GCN (GraphDTA) forward for MI455X / gfx1250.
// Dense layers: f16 WMMA (v_wmma_f32_16x16x32_f16) with f32 accumulate.
// Operands copied to f16 workspace with K padded to 32 (no K-tails, aligned
// rows); weights also transposed to [N,Kp]. GEMM uses LDS double-buffering
// with register-staged prefetch: global->VGPR after barrier, VGPR->LDS after
// the WMMAs, so VMEM latency hides behind matrix math.
#include <hip/hip_runtime.h>
#include <hip/hip_bf16.h>
#include <math.h>

typedef __attribute__((ext_vector_type(16))) _Float16 v16h;
typedef __attribute__((ext_vector_type(8)))  _Float16 v8h;
typedef __attribute__((ext_vector_type(4)))  _Float16 v4h;
typedef __attribute__((ext_vector_type(8)))  float    v8f;

// ------------------------------------------------- f32 [M,K] -> f16 [M,Kp]
__global__ void cvt_pad_f16(const float* __restrict__ in, _Float16* __restrict__ out,
                            int M, int K, int Kp) {
  int i4 = (blockIdx.x * blockDim.x + threadIdx.x) * 4;
  if (i4 >= M * Kp) return;
  int r = i4 / Kp, c = i4 % Kp;          // Kp % 4 == 0: quad never straddles rows
  v4h h;
#pragma unroll
  for (int j = 0; j < 4; ++j)
    h[j] = (c + j < K) ? (_Float16)in[(size_t)r * K + c + j] : (_Float16)0.0f;
  *(v4h*)(out + i4) = h;
}

// f32 [K,N] -> f16 [N,Kp] transpose+pad, tiled 32x32 via LDS
__global__ __launch_bounds__(256) void cvt_t_f32_f16(
    const float* __restrict__ in, _Float16* __restrict__ out,
    int K, int N, int Kp) {
  __shared__ float tile[32][33];
  int kt = blockIdx.y * 32, nt = blockIdx.x * 32;
  int tx = threadIdx.x & 31, ty = threadIdx.x >> 5;   // 32 x 8
  for (int i = ty; i < 32; i += 8) {
    int k = kt + i, n = nt + tx;
    tile[i][tx] = (k < K && n < N) ? in[(size_t)k * N + n] : 0.0f;
  }
  __syncthreads();
  for (int i = ty; i < 32; i += 8) {
    int n = nt + i, k = kt + tx;                      // k < Kp by grid construction
    if (n < N) out[(size_t)n * Kp + k] = (_Float16)tile[tx][i];
  }
}

// ---------------------------------------------------------------- WMMA GEMM
// C[M,N] = act(A16[M,Kp] @ Bt16[N,Kp]^T + bias); Kp % 32 == 0.
#define TBM 128
#define TBN 128
#define TBK 32
#define LSTR 40   // padded LDS row stride (f16); 40*2 % 16 == 0

// per-thread 128x32 tile slice: 2 x v8h (rows p*64 + tid/4, cols (tid&3)*8)
__device__ __forceinline__ void ld_tile(const _Float16* __restrict__ src, int Rows,
                                        int ldk, int r0, int k0, int tid, bool full,
                                        v8h out[2]) {
#pragma unroll
  for (int p = 0; p < 2; ++p) {
    int r = p * 64 + (tid >> 2);
    int c = (tid & 3) * 8;
    if (full || (r0 + r) < Rows) {
      out[p] = *(const v8h*)&src[(size_t)(r0 + r) * ldk + k0 + c];
    } else {
      v8h z = {};
      out[p] = z;
    }
  }
}
__device__ __forceinline__ void st_tile(_Float16* dst, int tid, const v8h in[2]) {
#pragma unroll
  for (int p = 0; p < 2; ++p) {
    int r = p * 64 + (tid >> 2);
    int c = (tid & 3) * 8;
    *(v8h*)&dst[r * LSTR + c] = in[p];
  }
}

__global__ __launch_bounds__(256) void gemm_wmma_f16(
    const _Float16* __restrict__ A, const _Float16* __restrict__ Bt,
    const float* __restrict__ bias, float* __restrict__ C,
    int M, int Kp, int N, int relu)
{
  __shared__ _Float16 sA[2][TBM * LSTR];
  __shared__ _Float16 sB[2][TBN * LSTR];   // sB[n*LSTR + k]

  const int tid  = threadIdx.x;
  const int lane = tid & 31;
  const int wave = tid >> 5;
  const int wm   = wave >> 2;             // 0..1  (64 rows each)
  const int wn   = wave & 3;              // 0..3  (32 cols each)
  const int m0   = blockIdx.y * TBM;
  const int n0   = blockIdx.x * TBN;
  const int frag = lane & 15;
  const int kb   = (lane >> 4) << 3;      // k-base 0 or 8 per ISA layout

  const bool fullA = (m0 + TBM <= M);
  const bool fullB = (n0 + TBN <= N);

  v8f acc[4][2];
#pragma unroll
  for (int mt = 0; mt < 4; ++mt)
#pragma unroll
    for (int nt = 0; nt < 2; ++nt)
#pragma unroll
      for (int r = 0; r < 8; ++r) acc[mt][nt][r] = 0.0f;

  // prologue: stage tile 0
  {
    v8h ta[2], tb[2];
    ld_tile(A,  M, Kp, m0, 0, tid, fullA, ta);
    ld_tile(Bt, N, Kp, n0, 0, tid, fullB, tb);
    st_tile(sA[0], tid, ta);
    st_tile(sB[0], tid, tb);
  }

  int cur = 0;
  for (int k0 = 0; k0 < Kp; k0 += TBK) {
    __syncthreads();   // staged tile visible; prior reads of other buffer done
    const bool hasNext = (k0 + TBK < Kp);
    v8h na[2], nb[2];
    if (hasNext) {                         // issue global loads early
      ld_tile(A,  M, Kp, m0, k0 + TBK, tid, fullA, na);
      ld_tile(Bt, N, Kp, n0, k0 + TBK, tid, fullB, nb);
    }

    const _Float16* cA = sA[cur];
    const _Float16* cB = sB[cur];
    v16h af[4];
#pragma unroll
    for (int mt = 0; mt < 4; ++mt) {
      int mr = wm * 64 + mt * 16 + frag;
      v8h lo = *(const v8h*)&cA[mr * LSTR + kb];        // K = kb..kb+7
      v8h hi = *(const v8h*)&cA[mr * LSTR + 16 + kb];   // K = 16+kb..23+kb
#pragma unroll
      for (int j = 0; j < 8; ++j) { af[mt][j] = lo[j]; af[mt][8 + j] = hi[j]; }
    }
    v16h bf[2];
#pragma unroll
    for (int nt = 0; nt < 2; ++nt) {
      int nr = wn * 32 + nt * 16 + frag;
      v8h lo = *(const v8h*)&cB[nr * LSTR + kb];
      v8h hi = *(const v8h*)&cB[nr * LSTR + 16 + kb];
#pragma unroll
      for (int j = 0; j < 8; ++j) { bf[nt][j] = lo[j]; bf[nt][8 + j] = hi[j]; }
    }
#pragma unroll
    for (int mt = 0; mt < 4; ++mt)
#pragma unroll
      for (int nt = 0; nt < 2; ++nt)
        acc[mt][nt] = __builtin_amdgcn_wmma_f32_16x16x32_f16(
            false, af[mt], false, bf[nt], (short)0, acc[mt][nt], false, false);

    if (hasNext) {                         // drain prefetch into other buffer
      st_tile(sA[cur ^ 1], tid, na);
      st_tile(sB[cur ^ 1], tid, nb);
    }
    cur ^= 1;
  }

  const int rbase = (lane >> 4) << 3;
  const int ncol  = lane & 15;
#pragma unroll
  for (int mt = 0; mt < 4; ++mt)
#pragma unroll
    for (int nt = 0; nt < 2; ++nt) {
      int gn = n0 + wn * 32 + nt * 16 + ncol;
      if (gn >= N) continue;
      float bv = bias ? bias[gn] : 0.0f;
#pragma unroll
      for (int r = 0; r < 8; ++r) {
        int gm = m0 + wm * 64 + mt * 16 + rbase + r;
        if (gm < M) {
          float v = acc[mt][nt][r] + bv;
          if (relu) v = fmaxf(v, 0.0f);
          C[(size_t)gm * N + gn] = v;
        }
      }
    }
}

// ---------------------------------------------------------------- helpers
__device__ __forceinline__ unsigned fenc(float x) {
  unsigned u = __float_as_uint(x);
  return (u & 0x80000000u) ? ~u : (u | 0x80000000u);   // monotone float->uint
}
__device__ __forceinline__ float fdec(unsigned u) {
  return (u & 0x80000000u) ? __uint_as_float(u & 0x7FFFFFFFu)
                           : __uint_as_float(~u);
}

__global__ void fill_u32(unsigned* __restrict__ p, unsigned v, int n) {
  int i = blockIdx.x * blockDim.x + threadIdx.x;
  if (i < n) p[i] = v;
}

// ---------------------------------------------------------------- conv1d
__global__ void conv1d_relu(const float* __restrict__ in, const float* __restrict__ w,
                            const float* __restrict__ b, float* __restrict__ out,
                            int Bn, int Cin, int Lin, int Cout) {
  int Lout = Lin - 7;
  int idx = blockIdx.x * blockDim.x + threadIdx.x;
  if (idx >= Bn * Cout * Lout) return;
  int l  = idx % Lout;
  int t  = idx / Lout;
  int co = t % Cout;
  int bb = t / Cout;
  const float* xin = in + (size_t)bb * Cin * Lin;
  const float* wr  = w + (size_t)co * Cin * 8;
  float acc = b[co];
  for (int ci = 0; ci < Cin; ++ci) {
    const float* xr = xin + ci * Lin + l;
    const float* wk = wr + ci * 8;
#pragma unroll
    for (int k = 0; k < 8; ++k) acc = fmaf(xr[k], wk[k], acc);
  }
  out[idx] = fmaxf(acc, 0.0f);
}

// First conv with embedding lookup fused: in[b,ci,l] = emb[ids[b,ci]][l], Lin=128
__global__ void conv1d_emb_relu(const int* __restrict__ ids, const float* __restrict__ emb,
                                const float* __restrict__ w, const float* __restrict__ b,
                                float* __restrict__ out, int Bn, int Cin, int Cout) {
  const int Lout = 121;  // 128 - 7
  int idx = blockIdx.x * blockDim.x + threadIdx.x;
  if (idx >= Bn * Cout * Lout) return;
  int l  = idx % Lout;
  int t  = idx / Lout;
  int co = t % Cout;
  int bb = t / Cout;
  const int*  idr = ids + (size_t)bb * Cin;
  const float* wr = w + (size_t)co * Cin * 8;
  float acc = b[co];
  for (int ci = 0; ci < Cin; ++ci) {
    const float* er = emb + (size_t)idr[ci] * 128 + l;
    const float* wk = wr + ci * 8;
#pragma unroll
    for (int k = 0; k < 8; ++k) acc = fmaf(er[k], wk[k], acc);
  }
  out[idx] = fmaxf(acc, 0.0f);
}

// maxpool1d(kernel=k) + flatten: in [B,C,Lin] -> out [B, C*Lp]
__global__ void maxpool_flat(const float* __restrict__ in, float* __restrict__ out,
                             int Bn, int C, int Lin, int k, int Lp) {
  int idx = blockIdx.x * blockDim.x + threadIdx.x;
  if (idx >= Bn * C * Lp) return;
  int j  = idx % Lp;
  int t  = idx / Lp;
  int c  = t % C;
  int bb = t / C;
  const float* p = in + ((size_t)bb * C + c) * Lin + j * k;
  float m = p[0];
  for (int q = 1; q < k; ++q) m = fmaxf(m, p[q]);
  out[idx] = m;
}

// ---------------------------------------------------------------- GAT pieces
__global__ void gat_scores(const float* __restrict__ h, const float* __restrict__ asrc,
                           const float* __restrict__ adst, float* __restrict__ ssrc,
                           float* __restrict__ sdst, int n, int H, int F) {
  int idx = blockIdx.x * blockDim.x + threadIdx.x;
  if (idx >= n * H) return;
  int hh = idx % H;
  int nn = idx / H;
  const float* hr = h + ((size_t)nn * H + hh) * F;
  const float* as = asrc + (size_t)hh * F;
  const float* ad = adst + (size_t)hh * F;
  float s1 = 0.f, s2 = 0.f;
  for (int f = 0; f < F; ++f) { s1 = fmaf(hr[f], as[f], s1); s2 = fmaf(hr[f], ad[f], s2); }
  ssrc[idx] = s1;
  sdst[idx] = s2;
}

__global__ void gat_edge1(const int* __restrict__ ei, int E, int Etot,
                          const float* __restrict__ ssrc, const float* __restrict__ sdst,
                          float* __restrict__ ebuf, unsigned* __restrict__ menc, int H) {
  int idx = blockIdx.x * blockDim.x + threadIdx.x;
  if (idx >= Etot * H) return;
  int hh = idx % H;
  int e  = idx / H;
  int s = (e < E) ? ei[e]     : (e - E);
  int d = (e < E) ? ei[E + e] : (e - E);
  float v = ssrc[s * H + hh] + sdst[d * H + hh];
  v = (v > 0.f) ? v : 0.2f * v;
  ebuf[idx] = v;
  atomicMax(&menc[d * H + hh], fenc(v));
}

__global__ void gat_edge2(const int* __restrict__ ei, int E, int Etot,
                          float* __restrict__ ebuf, const unsigned* __restrict__ menc,
                          float* __restrict__ den, int H) {
  int idx = blockIdx.x * blockDim.x + threadIdx.x;
  if (idx >= Etot * H) return;
  int hh = idx % H;
  int e  = idx / H;
  int d = (e < E) ? ei[E + e] : (e - E);
  float ex = __expf(ebuf[idx] - fdec(menc[d * H + hh]));
  ebuf[idx] = ex;
  atomicAdd(&den[d * H + hh], ex);
}

__global__ void gat_aggr(const int* __restrict__ ei, int E,
                         const float* __restrict__ hfeat, const float* __restrict__ ebuf,
                         const float* __restrict__ den, float* __restrict__ out,
                         int H, int F) {
  int e = blockIdx.x;
  int s = (e < E) ? ei[e]     : (e - E);
  int d = (e < E) ? ei[E + e] : (e - E);
  int HF = H * F;
  for (int f = threadIdx.x; f < HF; f += blockDim.x) {
    int hh = f / F;
    float alpha = ebuf[(size_t)e * H + hh] / (den[(size_t)d * H + hh] + 1e-16f);
    atomicAdd(&out[(size_t)d * HF + f], hfeat[(size_t)s * HF + f] * alpha);
  }
}

__global__ void bias_relu(float* __restrict__ x, const float* __restrict__ b,
                          int rows, int cols) {
  int idx = blockIdx.x * blockDim.x + threadIdx.x;
  if (idx >= rows * cols) return;
  x[idx] = fmaxf(x[idx] + b[idx % cols], 0.0f);
}

// ---------------------------------------------------------------- GCN pieces
__global__ void gcn_deg(const int* __restrict__ ei, int E, int Etot,
                        float* __restrict__ deg) {
  int e = blockIdx.x * blockDim.x + threadIdx.x;
  if (e >= Etot) return;
  int d = (e < E) ? ei[E + e] : (e - E);
  atomicAdd(&deg[d], 1.0f);
}
__global__ void gcn_dinv(const float* __restrict__ deg, float* __restrict__ dinv, int n) {
  int i = blockIdx.x * blockDim.x + threadIdx.x;
  if (i < n) dinv[i] = (deg[i] > 0.f) ? rsqrtf(deg[i]) : 0.0f;
}
__global__ void gcn_aggr(const int* __restrict__ ei, int E,
                         const float* __restrict__ h, const float* __restrict__ dinv,
                         float* __restrict__ out, int F) {
  int e = blockIdx.x;
  int s = (e < E) ? ei[e]     : (e - E);
  int d = (e < E) ? ei[E + e] : (e - E);
  float norm = dinv[s] * dinv[d];
  for (int f = threadIdx.x; f < F; f += blockDim.x)
    atomicAdd(&out[(size_t)d * F + f], h[(size_t)s * F + f] * norm);
}

// ---------------------------------------------------------------- pooling
__global__ void pool_node(const float* __restrict__ g, const int* __restrict__ batch,
                          unsigned* __restrict__ gmaxu, float* __restrict__ gsum,
                          int F) {
  int nn = blockIdx.x;
  int bg = batch[nn];
  for (int f = threadIdx.x; f < F; f += blockDim.x) {
    float v = g[(size_t)nn * F + f];
    atomicMax(&gmaxu[(size_t)bg * F + f], __float_as_uint(v));  // v >= 0 post-relu
    atomicAdd(&gsum[(size_t)bg * F + f], v);
  }
}
__global__ void pool_cnt(const int* __restrict__ batch, float* __restrict__ cnt, int n) {
  int i = blockIdx.x * blockDim.x + threadIdx.x;
  if (i < n) atomicAdd(&cnt[batch[i]], 1.0f);
}
__global__ void pool_combine(const unsigned* __restrict__ gmaxu,
                             const float* __restrict__ gsum,
                             const float* __restrict__ cnt,
                             float* __restrict__ gp, int Bn, int F) {
  int idx = blockIdx.x * blockDim.x + threadIdx.x;
  if (idx >= Bn * F) return;
  int b = idx / F, f = idx % F;
  gp[(size_t)b * 2 * F + f]     = __uint_as_float(gmaxu[idx]);
  gp[(size_t)b * 2 * F + F + f] = gsum[idx] / fmaxf(cnt[b], 1.0f);
}

__global__ void concat3(const float* __restrict__ a, const float* __restrict__ b,
                        const float* __restrict__ c, float* __restrict__ out, int Bn) {
  int idx = blockIdx.x * blockDim.x + threadIdx.x;
  if (idx >= Bn * 384) return;
  int row = idx / 384, col = idx % 384;
  float v = (col < 128) ? a[row * 128 + col]
          : (col < 256) ? b[row * 128 + col - 128]
                        : c[row * 128 + col - 256];
  out[idx] = v;
}

// ---------------------------------------------------------------- launcher
static inline void gemm(const float* A, const float* Bw, const float* bias, float* C,
                        int M, int K, int N, int relu,
                        _Float16* a16, _Float16* w16t, hipStream_t s) {
  int Kp = (K + 31) & ~31;
  int qa = (M * Kp + 3) / 4;
  cvt_pad_f16<<<(qa + 255) / 256, 256, 0, s>>>(A, a16, M, K, Kp);
  dim3 gt((N + 31) / 32, Kp / 32);
  cvt_t_f32_f16<<<gt, dim3(256), 0, s>>>(Bw, w16t, K, N, Kp);   // [N,Kp]
  dim3 g((N + TBN - 1) / TBN, (M + TBM - 1) / TBM);
  gemm_wmma_f16<<<g, dim3(256), 0, s>>>(a16, w16t, bias, C, M, Kp, N, relu);
}
static inline void fillf(float* p, float v, int n, hipStream_t s) {
  fill_u32<<<(n + 255) / 256, 256, 0, s>>>((unsigned*)p, __builtin_bit_cast(unsigned, v), n);
}
static inline void fillu(unsigned* p, unsigned v, int n, hipStream_t s) {
  fill_u32<<<(n + 255) / 256, 256, 0, s>>>(p, v, n);
}

extern "C" void kernel_launch(void* const* d_in, const int* in_sizes, int n_in,
                              void* d_out, int out_size, void* d_ws, size_t ws_size,
                              hipStream_t stream) {
  const float* x      = (const float*)d_in[0];
  const int*   ei     = (const int*)d_in[1];
  const int*   batch  = (const int*)d_in[2];
  const int*   target = (const int*)d_in[3];
  const int*   drugs  = (const int*)d_in[4];
  const float* P[48];
  for (int i = 5; i < n_in && i < 53; ++i) P[i - 5] = (const float*)d_in[i];

  const float *gat1_W = P[0], *gat1_as = P[1], *gat1_ad = P[2], *gat1_b = P[3];
  const float *gat2_W = P[4], *gat2_as = P[5], *gat2_ad = P[6], *gat2_b = P[7];
  const float *gcn_W  = P[8], *gcn_b   = P[9];
  const float *fcg1_W = P[10], *fcg1_b = P[11], *fcg2_W = P[12], *fcg2_b = P[13];
  const float *emb_xt = P[14];
  const float *ct1_W = P[15], *ct1_b = P[16], *ct2_W = P[17], *ct2_b = P[18];
  const float *ct3_W = P[19], *ct3_b = P[20], *ct4_W = P[21], *ct4_b = P[22];
  const float *f1xt_W = P[23], *f1xt_b = P[24], *f2xt_W = P[25], *f2xt_b = P[26];
  const float *emb_xd = P[27];
  const float *cd1_W = P[28], *cd1_b = P[29], *cd2_W = P[30], *cd2_b = P[31];
  const float *cd3_W = P[32], *cd3_b = P[33], *cd4_W = P[34], *cd4_b = P[35];
  const float *f1xd_W = P[36], *f1xd_b = P[37], *f2xd_W = P[38], *f2xd_b = P[39];
  const float *fc1_W = P[40], *fc1_b = P[41], *fc2_W = P[42], *fc2_b = P[43];
  const float *o_W = P[44], *o_b = P[45];

  const int NN = in_sizes[0] / 78;     // 10000 nodes
  const int E  = in_sizes[1] / 2;      // 40000 edges (self loops appended in-kernel)
  const int Et = E + NN;               // 50000
  const int Bn = in_sizes[3] / 1000;   // 256

  float* W = (float*)d_ws;
  const size_t REG = (size_t)NN * 3120;            // 31,200,000 floats
  float* bufA = W;                                  // h / CNN intermediates
  float* bufB = W + REG;                            // g outputs
  float* S    = W + 2 * REG;                        // small scratch
  float*    ssrc = S;                               // 100000
  float*    sdst = S + 100000;                      // 100000
  unsigned* menc = (unsigned*)(S + 200000);         // 100000
  float*    den  = S + 300000;                      // 100000
  float*    ebuf = S + 400000;                      // 500000
  float*    deg  = S + 900000;                      // 10000
  float*    dinv = S + 910000;                      // 10000
  float*    xds  = S + 920000;                      // 32768
  float*    xt   = S + 952768;                      // 32768
  float*    gpf  = S + 985536;                      // 32768
  float*    cat  = S + 1018304;                     // 98304
  float*    f1o  = S + 1116608;                     // 262144
  float*    f2o  = S + 1378752;                     // 131072
  _Float16* a16  = (_Float16*)(S + 1510400);        // 16,000,000 halves (8.0M floats)
  _Float16* w16  = (_Float16*)(S + 9510400);        // 6,500,000 halves  (3.25M floats)
  // total ws usage: ~75.2M floats (~301 MB)

  const int T = 256;
#define NB(n) (((n) + T - 1) / T)

  // ================= drug SMILES CNN branch (bufA scratch) =================
  {
    float* c1 = bufA;                 // [B,25,121]
    float* c2 = bufA + 774400;        // [B,50,114]
    float* c3 = bufA + 2233600;       // [B,75,107]
    float* c4 = bufA + 4288000;       // [B,100,100]
    float* pl = bufA + 6848000;       // [B,5000]
    float* fo = bufA + 8128000;       // [B,1024]
    conv1d_emb_relu<<<NB(Bn*25*121), T, 0, stream>>>(drugs, emb_xd, cd1_W, cd1_b, c1, Bn, 100, 25);
    conv1d_relu<<<NB(Bn*50*114), T, 0, stream>>>(c1, cd2_W, cd2_b, c2, Bn, 25, 121, 50);
    conv1d_relu<<<NB(Bn*75*107), T, 0, stream>>>(c2, cd3_W, cd3_b, c3, Bn, 50, 114, 75);
    conv1d_relu<<<NB(Bn*100*100), T, 0, stream>>>(c3, cd4_W, cd4_b, c4, Bn, 75, 107, 100);
    maxpool_flat<<<NB(Bn*100*50), T, 0, stream>>>(c4, pl, Bn, 100, 100, 2, 50);
    gemm(pl, f1xd_W, f1xd_b, fo, Bn, 5000, 1024, 1, a16, w16, stream);
    gemm(fo, f2xd_W, f2xd_b, xds, Bn, 1024, 128, 0, a16, w16, stream);
  }

  // ================= protein CNN branch (bufA scratch, reused) =============
  {
    float* c1 = bufA;                 // [B,32,121]
    float* c2 = bufA + 991232;        // [B,64,114]
    float* c3 = bufA + 2859008;       // [B,96,107]
    float* c4 = bufA + 5488640;       // [B,128,100]
    float* pl = bufA + 8765440;       // [B,4224]
    float* fo = bufA + 9846784;       // [B,1024]
    conv1d_emb_relu<<<NB(Bn*32*121), T, 0, stream>>>(target, emb_xt, ct1_W, ct1_b, c1, Bn, 1000, 32);
    conv1d_relu<<<NB(Bn*64*114), T, 0, stream>>>(c1, ct2_W, ct2_b, c2, Bn, 32, 121, 64);
    conv1d_relu<<<NB(Bn*96*107), T, 0, stream>>>(c2, ct3_W, ct3_b, c3, Bn, 64, 114, 96);
    conv1d_relu<<<NB(Bn*128*100), T, 0, stream>>>(c3, ct4_W, ct4_b, c4, Bn, 96, 107, 128);
    maxpool_flat<<<NB(Bn*128*33), T, 0, stream>>>(c4, pl, Bn, 128, 100, 3, 33);
    gemm(pl, f1xt_W, f1xt_b, fo, Bn, 4224, 1024, 1, a16, w16, stream);
    gemm(fo, f2xt_W, f2xt_b, xt, Bn, 1024, 128, 0, a16, w16, stream);
  }

  // ================= graph branch: GAT(10,78) -> GAT(2,780) -> GCN =========
  // --- GAT1 ---
  {
    const int H = 10, F = 78, HF = 780;
    gemm(x, gat1_W, nullptr, bufA, NN, 78, HF, 0, a16, w16, stream);   // h1
    gat_scores<<<NB(NN*H), T, 0, stream>>>(bufA, gat1_as, gat1_ad, ssrc, sdst, NN, H, F);
    fillu(menc, 0x007FFFFFu, NN*H, stream);                            // fenc(-inf)
    fillf(den, 0.0f, NN*H, stream);
    gat_edge1<<<NB(Et*H), T, 0, stream>>>(ei, E, Et, ssrc, sdst, ebuf, menc, H);
    gat_edge2<<<NB(Et*H), T, 0, stream>>>(ei, E, Et, ebuf, menc, den, H);
    fillf(bufB, 0.0f, NN*HF, stream);
    gat_aggr<<<Et, T, 0, stream>>>(ei, E, bufA, ebuf, den, bufB, H, F);
    bias_relu<<<NB(NN*HF), T, 0, stream>>>(bufB, gat1_b, NN, HF);      // g1 in bufB
  }
  // --- GAT2 ---
  {
    const int H = 2, F = 780, HF = 1560;
    gemm(bufB, gat2_W, nullptr, bufA, NN, 780, HF, 0, a16, w16, stream); // h2
    gat_scores<<<NB(NN*H), T, 0, stream>>>(bufA, gat2_as, gat2_ad, ssrc, sdst, NN, H, F);
    fillu(menc, 0x007FFFFFu, NN*H, stream);
    fillf(den, 0.0f, NN*H, stream);
    gat_edge1<<<NB(Et*H), T, 0, stream>>>(ei, E, Et, ssrc, sdst, ebuf, menc, H);
    gat_edge2<<<NB(Et*H), T, 0, stream>>>(ei, E, Et, ebuf, menc, den, H);
    fillf(bufB, 0.0f, NN*HF, stream);                                  // g1 dead now
    gat_aggr<<<Et, T, 0, stream>>>(ei, E, bufA, ebuf, den, bufB, H, F);
    bias_relu<<<NB(NN*HF), T, 0, stream>>>(bufB, gat2_b, NN, HF);      // g2 in bufB
  }
  // --- GCN ---
  {
    const int F = 3120;
    gemm(bufB, gcn_W, nullptr, bufA, NN, 1560, F, 0, a16, w16, stream); // h3
    fillf(deg, 0.0f, NN, stream);
    gcn_deg<<<NB(Et), T, 0, stream>>>(ei, E, Et, deg);
    gcn_dinv<<<NB(NN), T, 0, stream>>>(deg, dinv, NN);
    fillf(bufB, 0.0f, (int)((size_t)NN*F), stream);
    gcn_aggr<<<Et, T, 0, stream>>>(ei, E, bufA, dinv, bufB, F);
    bias_relu<<<NB(NN*F), T, 0, stream>>>(bufB, gcn_b, NN, F);          // g3 in bufB
  }
  // --- global max+mean pooling + graph FCs (bufA scratch) ---
  {
    const int F = 3120;
    unsigned* gmaxu = (unsigned*)bufA;                 // [B,3120]
    float*    gsum  = bufA + 798720;                   // [B,3120]
    float*    cnt   = bufA + 1597440;                  // [B]
    float*    gp    = bufA + 1597696;                  // [B,6240]
    float*    fo    = bufA + 3195136;                  // [B,1024]
    fillu(gmaxu, 0u, Bn*F, stream);
    fillf(gsum, 0.0f, Bn*F, stream);
    fillf(cnt, 0.0f, Bn, stream);
    pool_node<<<NN, T, 0, stream>>>(bufB, batch, gmaxu, gsum, F);
    pool_cnt<<<NB(NN), T, 0, stream>>>(batch, cnt, NN);
    pool_combine<<<NB(Bn*F), T, 0, stream>>>(gmaxu, gsum, cnt, gp, Bn, F);
    gemm(gp, fcg1_W, fcg1_b, fo, Bn, 6240, 1024, 1, a16, w16, stream);
    gemm(fo, fcg2_W, fcg2_b, gpf, Bn, 1024, 128, 0, a16, w16, stream);
  }

  // ================= fusion head =================
  concat3<<<NB(Bn*384), T, 0, stream>>>(xds, gpf, xt, cat, Bn);
  gemm(cat, fc1_W, fc1_b, f1o, Bn, 384, 1024, 1, a16, w16, stream);
  gemm(f1o, fc2_W, fc2_b, f2o, Bn, 1024, 512, 1, a16, w16, stream);
  gemm(f2o, o_W, o_b, (float*)d_out, Bn, 512, 1, 0, a16, w16, stream);
#undef NB
}
